// Contextual_Attention_Module_66159676228280
// MI455X (gfx1250) — matrix operand order repaired and checked by reading, not compiled, not run on hardware
//
#include <hip/hip_runtime.h>
#include <hip/hip_bf16.h>

// ---------------------------------------------------------------------------
// Contextual attention module for MI455X (gfx1250, wave32, WMMA).
// Pipeline per batch (B=4, C=128, H=W=128, Hs=Ws=64, L=4096, P=C*9=1152):
//   1) prep: wn (normalized bg patches, f16), fp (fg patches, f16),
//      rawT (kernel-4/stride-2 bg patches, transposed, f16)
//   2) GEMM1: Z[q,l] = fp . wn^T           (4096x1152x4096, f16->f32 WMMA)
//   3) fuse1/fuse2: diagonal band sums on flattened indices (matches ref)
//   4) row softmax (over l) -> attT f16
//   5) GEMM2: contrib = rawT . attT^T      (2048x4096x4096)
//   6) fold (stride-2 overlap-add, /4)
//   7) conv3x3+ELU twice via im2col + GEMM (128x1152x16384)
// All GEMMs share one v_wmma_f32_16x16x32_f16 tiled kernel with
// double-buffered GLOBAL_LOAD_ASYNC_TO_LDS_B128 staging (ASYNCcnt-fenced).
// ---------------------------------------------------------------------------

typedef __attribute__((ext_vector_type(16))) _Float16 v16h;
typedef __attribute__((ext_vector_type(8)))  _Float16 v8h;
typedef __attribute__((ext_vector_type(8)))  float    v8f;

#define BK 32

// Async global->LDS copy of 16 bytes. LDS byte address = low 32 bits of the
// flat shared-memory address.
__device__ __forceinline__ void async_copy_b128(const _Float16* g, uint32_t lds_off)
{
    asm volatile("global_load_async_to_lds_b128 %0, %1, off"
                 :: "v"(lds_off), "v"((unsigned long long)(uintptr_t)g)
                 : "memory");
}
__device__ __forceinline__ void wait_async0()
{
    asm volatile("s_wait_asynccnt 0x0" ::: "memory");
}

// ---------------------------------------------------------------------------
// Generic GEMM: C[M x N] = A[M x K] * Bt[N x K]^T ; A,Bt f16 row-major,
// C f32 row-major. M,N multiples of 128, K multiple of 32.
// Block = 256 threads (8 waves). Block tile 128x128. Wave tile 32x64
// (2x4 WMMA 16x16 tiles). Double-buffered async-staged 128x32 LDS tiles.
// ---------------------------------------------------------------------------
__global__ __launch_bounds__(256) void gemm_f16f32(
    const _Float16* __restrict__ A,
    const _Float16* __restrict__ Bt,
    float* __restrict__ C,
    int N, int K)
{
    __shared__ __align__(16) _Float16 lA[2][128 * BK];
    __shared__ __align__(16) _Float16 lB[2][128 * BK];

    const int tid  = threadIdx.x;
    const int lane = tid & 31;
    const int wid  = tid >> 5;
    const int wm   = wid & 3;        // 0..3 : 32-row strip
    const int wn   = wid >> 2;       // 0..1 : 64-col strip
    const int m0   = blockIdx.y * 128;
    const int n0   = blockIdx.x * 128;

    // staging map: each thread moves 16 contiguous f16 (two async b128 ops)
    const int ldRow  = tid >> 1;          // 0..127
    const int ldKoff = (tid & 1) * 16;    // 0 or 16

    v8f acc[2][4];
#pragma unroll
    for (int i = 0; i < 2; ++i)
#pragma unroll
        for (int j = 0; j < 4; ++j) acc[i][j] = v8f{};

    union V16 { v16h v; v8h h[2]; };

    const int nt = K / BK;

    auto stage = [&](int p, int kk) {
        const _Float16* gA = A  + (size_t)(m0 + ldRow) * K + kk + ldKoff;
        const _Float16* gB = Bt + (size_t)(n0 + ldRow) * K + kk + ldKoff;
        const uint32_t sA = (uint32_t)(uintptr_t)&lA[p][ldRow * BK + ldKoff];
        const uint32_t sB = (uint32_t)(uintptr_t)&lB[p][ldRow * BK + ldKoff];
        async_copy_b128(gA,     sA);
        async_copy_b128(gA + 8, sA + 16);
        async_copy_b128(gB,     sB);
        async_copy_b128(gB + 8, sB + 16);
        // warm L2 one tile further ahead
        if (kk + BK < K) {
            __builtin_prefetch(gA + BK, 0, 0);
            __builtin_prefetch(gB + BK, 0, 0);
        }
    };

    // prologue: fill buffer 0
    stage(0, 0);
    wait_async0();
    __syncthreads();

    for (int it = 0; it < nt; ++it) {
        const int p = it & 1;
        // kick off next tile into the other buffer (all waves finished
        // reading it at the previous barrier)
        if (it + 1 < nt) stage(p ^ 1, (it + 1) * BK);

        // A fragments: 16-bit A 16x32 layout — lanes 0-15: row=lane,
        // K 0-7 & 16-23 ; lanes 16-31: row=lane-16, K 8-15 & 24-31.
        const int arow = wm * 32 + (lane & 15);
        const int ak0  = (lane < 16) ? 0 : 8;
        V16 afrag[2];
#pragma unroll
        for (int i = 0; i < 2; ++i) {
            const _Float16* pa = &lA[p][(arow + i * 16) * BK + ak0];
            afrag[i].h[0] = *(const v8h*)pa;
            afrag[i].h[1] = *(const v8h*)(pa + 16);
        }
        // B fragments: lanes 0-15: col=lane, K 0-7 & 16-23 ; lanes 16-31:
        // col=lane-16, K 8-15 & 24-31 (same K order as A).
        const int bcol = wn * 64 + (lane & 15);
        const int bk0  = (lane < 16) ? 0 : 8;
        V16 bfrag[4];
#pragma unroll
        for (int j = 0; j < 4; ++j) {
            const _Float16* pb = &lB[p][(bcol + j * 16) * BK + bk0];
            bfrag[j].h[0] = *(const v8h*)pb;
            bfrag[j].h[1] = *(const v8h*)(pb + 16);
        }
#pragma unroll
        for (int i = 0; i < 2; ++i)
#pragma unroll
            for (int j = 0; j < 4; ++j)
                acc[i][j] = __builtin_amdgcn_wmma_f32_16x16x32_f16(
                    false, afrag[i].v, false, bfrag[j].v,
                    (short)0, acc[i][j], false, false);

        // next buffer fully landed in LDS + all waves done reading this one
        if (it + 1 < nt) wait_async0();
        __syncthreads();
    }

    // C/D layout: lanes 0-15 rows 0-7 (VGPR r -> row r), lanes 16-31 rows 8-15
    const int crow0 = m0 + wm * 32 + ((lane < 16) ? 0 : 8);
    const int ccol0 = n0 + wn * 64 + (lane & 15);
#pragma unroll
    for (int i = 0; i < 2; ++i)
#pragma unroll
        for (int j = 0; j < 4; ++j) {
            float* cp = C + (size_t)(crow0 + i * 16) * N + ccol0 + j * 16;
#pragma unroll
            for (int r = 0; r < 8; ++r) cp[(size_t)r * N] = acc[i][j][r];
        }
}

// ---------------------------------------------------------------------------
// Prep kernels (per batch image, C=128, H=W=128)
// ---------------------------------------------------------------------------

// wn[q][k] : normalized 3x3 patches of bs = b[:, ::2, ::2]; one block per q.
__global__ __launch_bounds__(256) void prep_wn_kernel(
    const float* __restrict__ b, _Float16* __restrict__ wn)
{
    __shared__ float patch[1152];
    __shared__ float red[256];
    const int q = blockIdx.x, tid = threadIdx.x;
    const int h = q >> 6, w = q & 63;
    float ss = 0.f;
    for (int k = tid; k < 1152; k += 256) {
        int c = k / 9, rem = k - c * 9;
        int dy = rem / 3, dx = rem - dy * 3;
        int y = h + dy - 1, x = w + dx - 1;
        float v = 0.f;
        if ((unsigned)y < 64u && (unsigned)x < 64u)
            v = b[c * 16384 + (2 * y) * 128 + 2 * x];
        patch[k] = v;
        ss += v * v;
    }
    red[tid] = ss; __syncthreads();
    for (int s = 128; s > 0; s >>= 1) {
        if (tid < s) red[tid] += red[tid + s];
        __syncthreads();
    }
    const float inv = 1.f / fmaxf(sqrtf(red[0]), 1e-4f);
    for (int k = tid; k < 1152; k += 256)
        wn[(size_t)q * 1152 + k] = (_Float16)(patch[k] * inv);
}

// fp[q][k] : 3x3 patches of fs = f[:, ::2, ::2]
__global__ __launch_bounds__(256) void prep_fp_kernel(
    const float* __restrict__ f, _Float16* __restrict__ fp)
{
    const int q = blockIdx.x;
    const int h = q >> 6, w = q & 63;
    for (int k = threadIdx.x; k < 1152; k += 256) {
        int c = k / 9, rem = k - c * 9;
        int dy = rem / 3, dx = rem - dy * 3;
        int y = h + dy - 1, x = w + dx - 1;
        float v = ((unsigned)y < 64u && (unsigned)x < 64u)
                      ? f[c * 16384 + (2 * y) * 128 + 2 * x] : 0.f;
        fp[(size_t)q * 1152 + k] = (_Float16)v;
    }
}

// rawT[m=c*16+i*4+j][l=h*64+w] = b[c, 2h+i-1, 2w+j-1] (k=4, s=2, pad 1)
__global__ __launch_bounds__(256) void prep_rawT_kernel(
    const float* __restrict__ b, _Float16* __restrict__ rawT)
{
    const int m = blockIdx.x;
    const int c = m >> 4, i = (m >> 2) & 3, j = m & 3;
    for (int l = threadIdx.x; l < 4096; l += 256) {
        int h = l >> 6, w = l & 63;
        int y = 2 * h + i - 1, x = 2 * w + j - 1;
        float v = ((unsigned)y < 128u && (unsigned)x < 128u)
                      ? b[c * 16384 + y * 128 + x] : 0.f;
        rawT[(size_t)m * 4096 + l] = (_Float16)v;
    }
}

// ---------------------------------------------------------------------------
// Diagonal band fusion (on flattened 4096-indices, matching reference)
// ---------------------------------------------------------------------------
__global__ __launch_bounds__(256) void fuse1_kernel(
    const float* __restrict__ Z, float* __restrict__ T)
{
    const int q = blockIdx.x;
    for (int l = threadIdx.x; l < 4096; l += 256) {
        float s = 0.f;
#pragma unroll
        for (int d = -1; d <= 1; ++d) {
            int qq = q + d, ll = l + d;
            if ((unsigned)qq < 4096u && (unsigned)ll < 4096u)
                s += Z[(size_t)qq * 4096 + ll];
        }
        T[(size_t)q * 4096 + l] = s;
    }
}

__device__ inline int wshift(int idx, int d)
{
    int wm = ((idx & 63) << 6) + (idx >> 6) + d;   // to w-major, shift
    if ((unsigned)wm >= 4096u) return -1;
    return ((wm & 63) << 6) + (wm >> 6);           // back to h-major
}

__global__ __launch_bounds__(256) void fuse2_kernel(
    const float* __restrict__ T, float* __restrict__ Z)
{
    const int q = blockIdx.x;
    for (int l = threadIdx.x; l < 4096; l += 256) {
        float s = 0.f;
#pragma unroll
        for (int d = -1; d <= 1; ++d) {
            int qq = wshift(q, d), ll = wshift(l, d);
            if (qq >= 0 && ll >= 0) s += T[(size_t)qq * 4096 + ll];
        }
        Z[(size_t)q * 4096 + l] = s;
    }
}

// ---------------------------------------------------------------------------
// Row softmax: attT[q][l] = softmax_l(10 * Z[q][l]) (Z is yi^T so this is
// the reference softmax over axis=0), output f16 (it is a GEMM B operand).
// ---------------------------------------------------------------------------
__global__ __launch_bounds__(256) void softmax_row_kernel(
    const float* __restrict__ Z, _Float16* __restrict__ attT)
{
    __shared__ float red[256];
    const int q = blockIdx.x, tid = threadIdx.x;
    const float* row = Z + (size_t)q * 4096;

    float mx = -3.4e38f;
    for (int l = tid; l < 4096; l += 256) mx = fmaxf(mx, row[l]);
    red[tid] = mx; __syncthreads();
    for (int s = 128; s > 0; s >>= 1) {
        if (tid < s) red[tid] = fmaxf(red[tid], red[tid + s]);
        __syncthreads();
    }
    mx = red[0]; __syncthreads();

    float sum = 0.f;
    for (int l = tid; l < 4096; l += 256) sum += __expf(10.f * (row[l] - mx));
    red[tid] = sum; __syncthreads();
    for (int s = 128; s > 0; s >>= 1) {
        if (tid < s) red[tid] += red[tid + s];
        __syncthreads();
    }
    const float inv = 1.f / red[0];
    for (int l = tid; l < 4096; l += 256)
        attT[(size_t)q * 4096 + l] =
            (_Float16)(__expf(10.f * (row[l] - mx)) * inv);
}

// ---------------------------------------------------------------------------
// Stride-2 overlap-add fold: y0[c,Y,X] = 1/4 * sum contrib[c,i,j,h,w]
// with Y+1 = i+2h, X+1 = j+2w.
// ---------------------------------------------------------------------------
__global__ __launch_bounds__(128) void fold_kernel(
    const float* __restrict__ contrib, float* __restrict__ y0)
{
    const int c = blockIdx.y, Y = blockIdx.x, X = threadIdx.x;
    const int i0 = (Y + 1) & 1, j0 = (X + 1) & 1;
    float s = 0.f;
    for (int i = i0; i < 4; i += 2) {
        int h = (Y + 1 - i) >> 1;
        if ((unsigned)h >= 64u) continue;
        for (int j = j0; j < 4; j += 2) {
            int w = (X + 1 - j) >> 1;
            if ((unsigned)w < 64u) {
                int m = c * 16 + i * 4 + j;
                s += contrib[(size_t)m * 4096 + h * 64 + w];
            }
        }
    }
    y0[c * 16384 + Y * 128 + X] = 0.25f * s;
}

// ---------------------------------------------------------------------------
// im2col for 3x3 SAME conv: cols[pix][c*9+dy*3+dx] (f16)
// ---------------------------------------------------------------------------
__global__ __launch_bounds__(288) void im2col_kernel(
    const float* __restrict__ x, _Float16* __restrict__ cols)
{
    const int pix = blockIdx.x;
    const int y = pix >> 7, xx = pix & 127;
    for (int k = threadIdx.x; k < 1152; k += 288) {
        int c = k / 9, rem = k - c * 9;
        int dy = rem / 3, dx = rem - dy * 3;
        int yy = y + dy - 1, xc = xx + dx - 1;
        float v = ((unsigned)yy < 128u && (unsigned)xc < 128u)
                      ? x[c * 16384 + yy * 128 + xc] : 0.f;
        cols[(size_t)pix * 1152 + k] = (_Float16)v;
    }
}

__global__ __launch_bounds__(256) void bias_elu_kernel(
    const float* __restrict__ conv, const float* __restrict__ bias,
    float* __restrict__ out)
{
    const int c = blockIdx.y;
    const int pix = blockIdx.x * 256 + threadIdx.x;
    float v = conv[(size_t)c * 16384 + pix] + bias[c];
    out[(size_t)c * 16384 + pix] = (v > 0.f) ? v : (__expf(v) - 1.f);
}

__global__ __launch_bounds__(256) void f32_to_f16_kernel(
    const float* __restrict__ src, _Float16* __restrict__ dst, int n)
{
    int i = blockIdx.x * 256 + threadIdx.x;
    if (i < n) dst[i] = (_Float16)src[i];
}

// ---------------------------------------------------------------------------
// Workspace layout (bytes, per-batch regions reused; aliases are safe by the
// launch ordering: yi free after softmax -> cols; t free after fuse2 ->
// contrib; attT free after GEMM2 -> act0/act1/convbuf).
// ---------------------------------------------------------------------------
static constexpr size_t SZ_WN    = 4096ull * 1152 * 2;   // 9.44 MB
static constexpr size_t SZ_RAWT  = 2048ull * 4096 * 2;   // 16.78 MB
static constexpr size_t SZ_YI    = 4096ull * 4096 * 4;   // 67.1 MB
static constexpr size_t SZ_ATT   = 4096ull * 4096 * 2;   // 33.55 MB
static constexpr size_t SZ_WH    = 128ull * 1152 * 2;

static constexpr size_t OFF_WN    = 0;
static constexpr size_t OFF_FP    = OFF_WN   + SZ_WN;
static constexpr size_t OFF_RAWT  = OFF_FP   + SZ_WN;
static constexpr size_t OFF_YI    = OFF_RAWT + SZ_RAWT;
static constexpr size_t OFF_T     = OFF_YI   + SZ_YI;
static constexpr size_t OFF_ATT   = OFF_T    + SZ_YI;
static constexpr size_t OFF_W1H   = OFF_ATT  + SZ_ATT;
static constexpr size_t OFF_W2H   = OFF_W1H  + SZ_WH;
// aliases
static constexpr size_t OFF_COLS    = OFF_YI;                    // 37.75 MB
static constexpr size_t OFF_CONTRIB = OFF_T;                     // 33.55 MB
static constexpr size_t OFF_ACT0    = OFF_ATT;                   // 8.39 MB
static constexpr size_t OFF_ACT1    = OFF_ATT + 8ull  * 1024 * 1024;
static constexpr size_t OFF_CONVB   = OFF_ATT + 16ull * 1024 * 1024;

extern "C" void kernel_launch(void* const* d_in, const int* in_sizes, int n_in,
                              void* d_out, int out_size, void* d_ws, size_t ws_size,
                              hipStream_t stream)
{
    (void)in_sizes; (void)n_in; (void)out_size; (void)ws_size;

    const float* f  = (const float*)d_in[0];
    const float* b  = (const float*)d_in[1];
    const float* w1 = (const float*)d_in[2];
    const float* b1 = (const float*)d_in[3];
    const float* w2 = (const float*)d_in[4];
    const float* b2 = (const float*)d_in[5];
    float* out = (float*)d_out;
    char* ws = (char*)d_ws;

    _Float16* wn      = (_Float16*)(ws + OFF_WN);
    _Float16* fp      = (_Float16*)(ws + OFF_FP);
    _Float16* rawT    = (_Float16*)(ws + OFF_RAWT);
    float*    yi      = (float*)   (ws + OFF_YI);
    float*    t       = (float*)   (ws + OFF_T);
    _Float16* attT    = (_Float16*)(ws + OFF_ATT);
    _Float16* w1h     = (_Float16*)(ws + OFF_W1H);
    _Float16* w2h     = (_Float16*)(ws + OFF_W2H);
    _Float16* cols    = (_Float16*)(ws + OFF_COLS);
    float*    contrib = (float*)   (ws + OFF_CONTRIB);
    float*    act0    = (float*)   (ws + OFF_ACT0);
    float*    act1    = (float*)   (ws + OFF_ACT1);
    float*    convbuf = (float*)   (ws + OFF_CONVB);

    f32_to_f16_kernel<<<(147456 + 255) / 256, 256, 0, stream>>>(w1, w1h, 147456);
    f32_to_f16_kernel<<<(147456 + 255) / 256, 256, 0, stream>>>(w2, w2h, 147456);

    for (int bi = 0; bi < 4; ++bi) {
        const float* fb = f + (size_t)bi * 128 * 128 * 128;
        const float* bb = b + (size_t)bi * 128 * 128 * 128;
        float*       ob = out + (size_t)bi * 128 * 128 * 128;

        prep_wn_kernel  <<<4096, 256, 0, stream>>>(bb, wn);
        prep_fp_kernel  <<<4096, 256, 0, stream>>>(fb, fp);
        prep_rawT_kernel<<<2048, 256, 0, stream>>>(bb, rawT);

        // Z = fp . wn^T  (M=4096, N=4096, K=1152)
        gemm_f16f32<<<dim3(32, 32), 256, 0, stream>>>(fp, wn, yi, 4096, 1152);

        fuse1_kernel<<<4096, 256, 0, stream>>>(yi, t);
        fuse2_kernel<<<4096, 256, 0, stream>>>(t, yi);
        softmax_row_kernel<<<4096, 256, 0, stream>>>(yi, attT);

        // contrib = rawT . attT^T  (M=2048, N=4096, K=4096)
        gemm_f16f32<<<dim3(32, 16), 256, 0, stream>>>(rawT, attT, contrib, 4096, 4096);

        fold_kernel<<<dim3(128, 128), 128, 0, stream>>>(contrib, act0);

        // conv1: M=128, N=16384, K=1152
        im2col_kernel<<<16384, 288, 0, stream>>>(act0, cols);
        gemm_f16f32<<<dim3(128, 1), 256, 0, stream>>>(w1h, cols, convbuf, 16384, 1152);
        bias_elu_kernel<<<dim3(64, 128), 256, 0, stream>>>(convbuf, b1, act1);

        // conv2
        im2col_kernel<<<16384, 288, 0, stream>>>(act1, cols);
        gemm_f16f32<<<dim3(128, 1), 256, 0, stream>>>(w2h, cols, convbuf, 16384, 1152);
        bias_elu_kernel<<<dim3(64, 128), 256, 0, stream>>>(convbuf, b2, ob);
    }
}
